// Encoder_v2_82721070121757
// MI455X (gfx1250) — matrix operand order repaired and checked by reading, not compiled, not run on hardware
//
#include <hip/hip_runtime.h>
#include <math.h>

// ---------------------------------------------------------------------------
// MI455X (gfx1250) implementation, round 3.
//  - All GEMM operands are bf16-resident; GEMMs use v_wmma_f32_16x16x32_bf16.
//  - Tile staging via global_load_async_to_lds_b128 (ASYNCcnt) with
//    double-buffered LDS, overlapping DMA with WMMA. Sync uint4 fallback if
//    the async builtins are unavailable.
//  - No bounds checks in staging: M % 64 == 0, K % 32 == 0 by construction,
//    and B buffers are padded to multiples of 128 rows (guarded writeback).
// ---------------------------------------------------------------------------

typedef __attribute__((ext_vector_type(16))) __bf16 v16bf;
typedef __attribute__((ext_vector_type(8)))  float  v8f;

#if __has_builtin(__builtin_amdgcn_global_load_async_to_lds_b128) && \
    __has_builtin(__builtin_amdgcn_s_wait_asynccnt)
#define HAVE_ASYNC 1
typedef int v4i __attribute__((vector_size(16)));   // builtin's pointee type
#endif

__device__ __forceinline__ unsigned short f2bf(float f) {
  union { float f; unsigned u; } x; x.f = f;
  unsigned r = x.u + 0x7FFFu + ((x.u >> 16) & 1u);   // round-to-nearest-even
  return (unsigned short)(r >> 16);
}

// ------------------------------ WMMA GEMM ----------------------------------
// acc = A[M,K](bf16, lda) * B(bf16, [N][K] rows, ldb)
// v = acc + bias; act; v += residual;  ->  outf (f32), out16 (bf16),
// out16t (bf16 transposed, [n][m], ldt).
#define BM 64
#define BN 128
#define BKK 32
#define LDS_PAD 40   // shorts per LDS row: 80B, 16B aligned, bank-staggered

__global__ __launch_bounds__(256)
void gemm_bf16_wmma(const unsigned short* __restrict__ A, int lda,
                    const unsigned short* __restrict__ B, int ldb,
                    const float* __restrict__ bias,
                    const float* __restrict__ residual,
                    float* __restrict__ outf, int ldc,
                    unsigned short* __restrict__ out16, int ld16,
                    unsigned short* __restrict__ out16t, int ldt,
                    int M, int N, int K, int act)
{
  __shared__ __align__(16) unsigned short As[2][BM * LDS_PAD];
  __shared__ __align__(16) unsigned short Bs[2][BN * LDS_PAD];

  const int tid  = threadIdx.x;
  const int lane = tid & 31;
  const int wave = tid >> 5;
  const int wm   = wave >> 2;            // 0..1 (M)
  const int wn   = wave & 3;             // 0..3 (N)
  const int blockM = blockIdx.y * BM;
  const int blockN = blockIdx.x * BN;

  v8f acc[2][2] = {};

  // stage one 64x32 A tile + 128x32 B tile into LDS buffer `buf`.
  // 16B chunk per thread for A (256 chunks), 2 chunks for B (512 chunks).
  auto stage = [&](int buf, int k0) {
    {
      int m = tid >> 2, c = (tid & 3) * 8;
      const unsigned short* src = A + (size_t)(blockM + m) * lda + k0 + c;
      unsigned short* dst = &As[buf][m * LDS_PAD + c];
#ifdef HAVE_ASYNC
      __builtin_amdgcn_global_load_async_to_lds_b128((v4i*)src, (v4i*)dst, 0, 0);
#else
      *(uint4*)dst = *(const uint4*)src;
#endif
    }
#pragma unroll
    for (int i0 = 0; i0 < 2; ++i0) {
      int i = tid + i0 * 256;
      int n = i >> 2, c = (i & 3) * 8;
      const unsigned short* src = B + (size_t)(blockN + n) * ldb + k0 + c;
      unsigned short* dst = &Bs[buf][n * LDS_PAD + c];
#ifdef HAVE_ASYNC
      __builtin_amdgcn_global_load_async_to_lds_b128((v4i*)src, (v4i*)dst, 0, 0);
#else
      *(uint4*)dst = *(const uint4*)src;
#endif
    }
  };

  const int nk = K >> 5;          // K is always a multiple of 32
  stage(0, 0);

  const int g = lane >> 4;        // half-wave group
  const int r = lane & 15;

  for (int s = 0; s < nk; ++s) {
    int cur = s & 1;
    if (s + 1 < nk) {
      stage(cur ^ 1, (s + 1) << 5);     // issue next tile (async: in flight)
#ifdef HAVE_ASYNC
      __builtin_amdgcn_s_wait_asynccnt(3);   // current tile's 3 ops complete
#endif
    } else {
#ifdef HAVE_ASYNC
      __builtin_amdgcn_s_wait_asynccnt(0);
#endif
    }
    __syncthreads();   // current buffer visible to all waves

    v16bf afrag[2], bfrag[2];
#pragma unroll
    for (int sm = 0; sm < 2; ++sm) {
      const unsigned short* base = &As[cur][(wm * 32 + sm * 16 + r) * LDS_PAD];
      union { uint4 q[2]; v16bf v; } u;
      u.q[0] = *(const uint4*)(base + g * 16);
      u.q[1] = *(const uint4*)(base + g * 16 + 8);
      afrag[sm] = u.v;
    }
#pragma unroll
    for (int sn = 0; sn < 2; ++sn) {
      const unsigned short* base =
          &Bs[cur][(wn * 32 + sn * 16 + r) * LDS_PAD + g * 16];
      union { uint4 q[2]; v16bf v; } u;
      u.q[0] = *(const uint4*)(base);
      u.q[1] = *(const uint4*)(base + 8);
      bfrag[sn] = u.v;
    }

#pragma unroll
    for (int sm = 0; sm < 2; ++sm)
#pragma unroll
      for (int sn = 0; sn < 2; ++sn)
        acc[sm][sn] = __builtin_amdgcn_wmma_f32_16x16x32_bf16(
            false, afrag[sm], false, bfrag[sn], (short)0, acc[sm][sn],
            false, false);

    __syncthreads();   // all reads done before this buffer is re-staged
  }

  // writeback: lane holds col n = base+(l%16), rows rr + 8*(l/16)
  const int cn = lane & 15;
#pragma unroll
  for (int sm = 0; sm < 2; ++sm)
#pragma unroll
    for (int sn = 0; sn < 2; ++sn) {
      int n = blockN + wn * 32 + sn * 16 + cn;
      if (n >= N) continue;
#pragma unroll
      for (int rr = 0; rr < 8; ++rr) {
        int m = blockM + wm * 32 + sm * 16 + g * 8 + rr;
        if (m >= M) continue;
        float v = acc[sm][sn][rr];
        if (bias) v += bias[n];
        if (act == 1) v = 0.5f * v * (1.f + erff(v * 0.70710678118f));
        if (residual) v += residual[(size_t)m * ldc + n];
        if (outf)   outf[(size_t)m * ldc + n] = v;
        if (out16)  out16[(size_t)m * ld16 + n] = f2bf(v);
        if (out16t) out16t[(size_t)n * ldt + m] = f2bf(v);
      }
    }
}

// ----------------- weight transpose + convert:  w[K][N] -> o[N][K] ---------
__global__ void wtrans_bf16(const float* __restrict__ w,
                            unsigned short* __restrict__ o, int K, int N)
{
  int i = blockIdx.x * 256 + threadIdx.x;
  if (i >= K * N) return;
  int n = i % N, k = i / N;
  o[(size_t)n * K + k] = f2bf(w[(size_t)k * N + n]);
}

// ------------------------------ LayerNorm ----------------------------------
__global__ __launch_bounds__(128)
void layernorm_rows(const float* __restrict__ x, const float* __restrict__ g,
                    const float* __restrict__ b, float* __restrict__ y,
                    unsigned short* __restrict__ y16, int rows, int C)
{
  int row  = blockIdx.x * 4 + (threadIdx.x >> 5);
  int lane = threadIdx.x & 31;
  if (row >= rows) return;
  const float* xr = x + (size_t)row * C;
  float s = 0.f;
  for (int c = lane; c < C; c += 32) s += xr[c];
  for (int o = 16; o; o >>= 1) s += __shfl_xor(s, o, 32);
  float mean = s / C;
  float v = 0.f;
  for (int c = lane; c < C; c += 32) { float d = xr[c] - mean; v += d * d; }
  for (int o = 16; o; o >>= 1) v += __shfl_xor(v, o, 32);
  float rstd = rsqrtf(v / C + 1e-5f);
  for (int c = lane; c < C; c += 32) {
    float val = (xr[c] - mean) * rstd * g[c] + b[c];
    if (y)   y[(size_t)row * C + c] = val;
    if (y16) y16[(size_t)row * C + c] = f2bf(val);
  }
}

// --------------- Softmax: f32 scores (in place exp) -> bf16 probs ----------
__global__ __launch_bounds__(128)
void softmax_rows(float* __restrict__ s, unsigned short* __restrict__ p16,
                  int rows, int cols, float scale)
{
  int row  = blockIdx.x * 4 + (threadIdx.x >> 5);
  int lane = threadIdx.x & 31;
  if (row >= rows) return;
  float* r = s + (size_t)row * cols;
  unsigned short* p = p16 + (size_t)row * cols;
  float mx = -3.4e38f;
  for (int c = lane; c < cols; c += 32) mx = fmaxf(mx, r[c] * scale);
  for (int o = 16; o; o >>= 1) mx = fmaxf(mx, __shfl_xor(mx, o, 32));
  float sum = 0.f;
  for (int c = lane; c < cols; c += 32) {
    float e = __expf(r[c] * scale - mx); r[c] = e; sum += e;
  }
  for (int o = 16; o; o >>= 1) sum += __shfl_xor(sum, o, 32);
  float inv = 1.f / sum;
  for (int c = lane; c < cols; c += 32) p[c] = f2bf(r[c] * inv);
}

// ---------------------- Neighborhood attention -----------------------------
// qkv: [L][3C] f32 (q|k|v), d = C/heads == 32 at every level. out: bf16.
__global__ __launch_bounds__(256)
void nat_attn(const float* __restrict__ qkv, const float* __restrict__ rpb,
              unsigned short* __restrict__ out16, int H, int W, int C,
              int heads)
{
  int idx = blockIdx.x * 256 + threadIdx.x;
  int L = H * W;
  if (idx >= L * heads) return;
  int hh = idx % heads;
  int t  = idx / heads;
  int y = t / W, x = t % W;
  int C3 = 3 * C;
  const float scale = 0.17677669529f;    // 32^-0.5

  float q[32];
  const float* qp = qkv + (size_t)t * C3 + hh * 32;
#pragma unroll
  for (int j = 0; j < 32; ++j) q[j] = qp[j] * scale;

  int sy = y - 3; if (sy < 0) sy = 0; if (sy > H - 7) sy = H - 7;
  int sx = x - 3; if (sx < 0) sx = 0; if (sx > W - 7) sx = W - 7;

  float lg[49];
  float mx = -3.4e38f;
  for (int i = 0; i < 7; ++i)
    for (int j2 = 0; j2 < 7; ++j2) {
      int ny = sy + i, nx = sx + j2;
      const float* kp = qkv + (size_t)(ny * W + nx) * C3 + C + hh * 32;
      float s = 0.f;
#pragma unroll
      for (int j = 0; j < 32; ++j) s += q[j] * kp[j];
      s += rpb[(hh * 13 + (ny - y + 6)) * 13 + (nx - x + 6)];
      lg[i * 7 + j2] = s;
      mx = fmaxf(mx, s);
    }
  float sum = 0.f;
  for (int n = 0; n < 49; ++n) { float e = __expf(lg[n] - mx); lg[n] = e; sum += e; }
  float inv = 1.f / sum;

  float o[32];
#pragma unroll
  for (int j = 0; j < 32; ++j) o[j] = 0.f;
  for (int i = 0; i < 7; ++i)
    for (int j2 = 0; j2 < 7; ++j2) {
      int ny = sy + i, nx = sx + j2;
      const float* vp = qkv + (size_t)(ny * W + nx) * C3 + 2 * C + hh * 32;
      float p = lg[i * 7 + j2] * inv;
#pragma unroll
      for (int j = 0; j < 32; ++j) o[j] += p * vp[j];
    }
  unsigned short* op = out16 + (size_t)t * C + hh * 32;
#pragma unroll
  for (int j = 0; j < 32; ++j) op[j] = f2bf(o[j]);
}

// ------------------- 3x3 stride-2 pad-1 convolution ------------------------
__global__ void conv3x3_s2(const float* __restrict__ in, int Cin, int Hin,
                           int Win, int in_nchw,
                           const float* __restrict__ w,
                           const float* __restrict__ bias,
                           float* __restrict__ out, int Cout)
{
  int Hout = Hin / 2, Wout = Win / 2;
  int idx = blockIdx.x * blockDim.x + threadIdx.x;
  int total = Hout * Wout * Cout;
  if (idx >= total) return;
  int co = idx % Cout;
  int x  = (idx / Cout) % Wout;
  int y  = idx / (Cout * Wout);
  float acc = bias ? bias[co] : 0.f;
  for (int ci = 0; ci < Cin; ++ci)
    for (int ky = 0; ky < 3; ++ky)
      for (int kx = 0; kx < 3; ++kx) {
        int iy = y * 2 + ky - 1, ix = x * 2 + kx - 1;
        if (iy < 0 || iy >= Hin || ix < 0 || ix >= Win) continue;
        float v = in_nchw ? in[((size_t)ci * Hin + iy) * Win + ix]
                          : in[((size_t)(iy * Win + ix)) * Cin + ci];
        acc += v * w[(((size_t)co * Cin + ci) * 3 + ky) * 3 + kx];
      }
  out[idx] = acc;
}

// ---------------------------------------------------------------------------
extern "C" void kernel_launch(void* const* d_in, const int* in_sizes, int n_in,
                              void* d_out, int out_size, void* d_ws,
                              size_t ws_size, hipStream_t stream)
{
  const int DEPTH[4]  = {2, 2, 4, 2};
  const int NHEADS[4] = {3, 6, 12, 24};

  // ---- input pointer walk (setup_inputs dict insertion order) ----
  int pidx = 0;
  auto nxt = [&]() { return (const float*)d_in[pidx++]; };
  const float* x1_in = nxt();
  const float* x2_in = nxt();
  const float *pw1 = nxt(), *pb1 = nxt(), *pw2 = nxt(), *pb2 = nxt();
  const float *png = nxt(), *pnb = nxt();

  struct Cross { const float *wq,*bq,*wk,*bk,*wv,*bv,*wp,*bp; };
  struct Blk   { const float *n1g,*n1b,*wqkv,*bqkv,*rpb,*wp,*bp,*n2g,*n2b,*w1,*b1,*w2,*b2; };
  struct Down  { const float *w,*g,*b; };
  Cross cross[4]; Blk blks[4][4]; Down down[3];
  for (int i = 0; i < 4; ++i) {
    cross[i] = { nxt(),nxt(),nxt(),nxt(),nxt(),nxt(),nxt(),nxt() };
    for (int bb = 0; bb < DEPTH[i]; ++bb)
      blks[i][bb] = { nxt(),nxt(),nxt(),nxt(),nxt(),nxt(),nxt(),
                      nxt(),nxt(),nxt(),nxt(),nxt(),nxt() };
    if (i < 3) down[i] = { nxt(), nxt(), nxt() };
  }
  const float *ong[4], *onb[4];
  for (int i = 0; i < 4; ++i) { ong[i] = nxt(); onb[i] = nxt(); }

  // ---- workspace carve (256B-aligned chunks) ----
  char* wsp = (char*)d_ws;
  auto carveB = [&](size_t bytes) {
    char* r = wsp; wsp += (bytes + 255) & ~(size_t)255; return r;
  };
  const size_t S = 1179648;   // max L*3C over levels (4096*288)

  float* xa   = (float*)carveB(S * 4);
  float* xb   = (float*)carveB(S * 4);
  float* ya   = (float*)carveB(S * 4);
  float* yb   = (float*)carveB(S * 4);
  float* qkvb = (float*)carveB(S * 4);
  float* cb   = (float*)carveB((size_t)128 * 128 * 48 * 4);
  float* sb   = (float*)carveB((size_t)512 * 4096 * 4);

  unsigned short* xa16 = (unsigned short*)carveB(S * 2);
  unsigned short* xb16 = (unsigned short*)carveB(S * 2);
  unsigned short* ya16 = (unsigned short*)carveB(S * 2);
  unsigned short* yb16 = (unsigned short*)carveB(S * 2);
  unsigned short* ln16 = (unsigned short*)carveB(S * 2);
  unsigned short* q16  = (unsigned short*)carveB(S * 2);
  unsigned short* k16  = (unsigned short*)carveB(S * 2);
  unsigned short* vt16 = (unsigned short*)carveB(S * 2);
  unsigned short* at16 = (unsigned short*)carveB(S * 2);
  unsigned short* h16  = (unsigned short*)carveB(S * 2);
  unsigned short* p16  = (unsigned short*)carveB((size_t)512 * 4096 * 2);

  // ---- one-time weight transpose+convert: w[K][N] -> wT[N][K] bf16,
  //      carved with N rounded up to 128 rows so B staging never reads OOB.
  auto carveW = [&](const float* w, int K, int N) {
    int rows = (N + 127) & ~127;
    unsigned short* o = (unsigned short*)carveB((size_t)rows * K * 2);
    int total = K * N;
    wtrans_bf16<<<dim3((total + 255) / 256), 256, 0, stream>>>(w, o, K, N);
    return o;
  };
  struct CrossW { unsigned short *wq,*wk,*wv,*wp; };
  struct BlkW   { unsigned short *wqkv,*wp,*w1,*w2; };
  CrossW crw[4]; BlkW blw[4][4];
  for (int lv = 0; lv < 4; ++lv) {
    int D = 96 << lv;
    crw[lv] = { carveW(cross[lv].wq, D, D), carveW(cross[lv].wk, D, D),
                carveW(cross[lv].wv, D, D), carveW(cross[lv].wp, D, D) };
    for (int bb = 0; bb < DEPTH[lv]; ++bb)
      blw[lv][bb] = { carveW(blks[lv][bb].wqkv, D, 3 * D),
                      carveW(blks[lv][bb].wp,   D, D),
                      carveW(blks[lv][bb].w1,   D, 3 * D),
                      carveW(blks[lv][bb].w2,   3 * D, D) };
  }

  // ---- launch helpers ----
  auto gemm = [&](const unsigned short* A, int lda, const unsigned short* B,
                  int ldb, const float* bias, const float* resid,
                  float* outf, int ldc, unsigned short* o16, int ld16,
                  unsigned short* o16t, int ldt, int M, int N, int K, int act) {
    dim3 grid((N + BN - 1) / BN, (M + BM - 1) / BM);
    gemm_bf16_wmma<<<grid, 256, 0, stream>>>(A, lda, B, ldb, bias, resid,
                                             outf, ldc, o16, ld16, o16t, ldt,
                                             M, N, K, act);
  };
  auto lnorm = [&](const float* x, const float* g, const float* b, float* y,
                   unsigned short* y16, int rows, int C) {
    layernorm_rows<<<dim3((rows + 3) / 4), 128, 0, stream>>>(x, g, b, y, y16,
                                                             rows, C);
  };

  // ---- patch tokenizer ----
  auto tok = [&](const float* x_in, float* dst, unsigned short* dst16) {
    conv3x3_s2<<<dim3((128 * 128 * 48 + 255) / 256), 256, 0, stream>>>(
        x_in, 3, 256, 256, 1, pw1, pb1, cb, 48);
    conv3x3_s2<<<dim3((64 * 64 * 96 + 255) / 256), 256, 0, stream>>>(
        cb, 48, 128, 128, 0, pw2, pb2, qkvb, 96);
    lnorm(qkvb, png, pnb, dst, dst16, 4096, 96);
  };
  tok(x1_in, xa, xa16);
  tok(x2_in, xb, xb16);

  // ---- cross differential attention ----
  auto cross_dif = [&](const unsigned short* xq16v,
                       const unsigned short* xkv16v, const float* xkvf,
                       const Cross& cp, const CrossW& cw, int heads,
                       float* dst, unsigned short* dst16, int L, int C) {
    gemm(xq16v,  C, cw.wq, C, cp.bq, nullptr, nullptr, 0, q16, C,
         nullptr, 0, L, C, C, 0);
    gemm(xkv16v, C, cw.wk, C, cp.bk, nullptr, nullptr, 0, k16, C,
         nullptr, 0, L, C, C, 0);
    gemm(xkv16v, C, cw.wv, C, cp.bv, nullptr, nullptr, 0, nullptr, 0,
         vt16, L, L, C, C, 0);   // V stored transposed: vt16[n][m], n over C
    int chunk = (L < 512) ? L : 512;
    for (int hh = 0; hh < heads; ++hh)
      for (int m0 = 0; m0 < L; m0 += chunk) {
        // scores[chunk,L] = q_h * k_h^T  (B rows = keys, 32 contiguous K)
        gemm(q16 + (size_t)m0 * C + hh * 32, C, k16 + hh * 32, C,
             nullptr, nullptr, sb, L, nullptr, 0, nullptr, 0,
             chunk, L, 32, 0);
        softmax_rows<<<dim3((chunk + 3) / 4), 128, 0, stream>>>(
            sb, p16, chunk, L, 0.17677669529f);
        // out_h = P * v_h  (B rows = vt16 rows = d-channels, K contiguous)
        gemm(p16, L, vt16 + (size_t)hh * 32 * L, L, nullptr, nullptr,
             nullptr, 0, at16 + (size_t)m0 * C + hh * 32, C, nullptr, 0,
             chunk, 32, L, 0);
      }
    // proj + bias + residual(b_ = xkv)
    gemm(at16, C, cw.wp, C, cp.bp, xkvf, dst, C, dst16, C, nullptr, 0,
         L, C, C, 0);
  };

  // ---- NAT transformer layer (in place on x / x16) ----
  auto nat_layer = [&](float* x, unsigned short* x16, const Blk& bp,
                       const BlkW& bw, int heads, int L, int C, int H, int W) {
    lnorm(x, bp.n1g, bp.n1b, nullptr, ln16, L, C);
    gemm(ln16, C, bw.wqkv, C, bp.bqkv, nullptr, qkvb, 3 * C, nullptr, 0,
         nullptr, 0, L, 3 * C, C, 0);
    nat_attn<<<dim3((L * heads + 255) / 256), 256, 0, stream>>>(
        qkvb, bp.rpb, at16, H, W, C, heads);
    gemm(at16, C, bw.wp, C, bp.bp, x, x, C, x16, C, nullptr, 0,
         L, C, C, 0);                          // + residual, mirror to bf16
    lnorm(x, bp.n2g, bp.n2b, nullptr, ln16, L, C);
    gemm(ln16, C, bw.w1, C, bp.b1, nullptr, nullptr, 0, h16, 3 * C,
         nullptr, 0, L, 3 * C, C, 1);          // gelu
    gemm(h16, 3 * C, bw.w2, 3 * C, bp.b2, x, x, C, x16, C, nullptr, 0,
         L, C, 3 * C, 0);                      // + residual, mirror to bf16
  };

  // ---- main level loop ----
  float* x1p = xa; float* x2p = xb; float* y1p = ya; float* y2p = yb;
  unsigned short* x1q = xa16; unsigned short* x2q = xb16;
  unsigned short* y1q = ya16; unsigned short* y2q = yb16;
  int H = 64, W = 64, C = 96;
  size_t out_off = 0;
  float* outp = (float*)d_out;

  for (int lv = 0; lv < 4; ++lv) {
    int L = H * W;
    int heads = NHEADS[lv];

    cross_dif(x1q, x2q, x2p, cross[lv], crw[lv], heads, y1p, y1q, L, C);
    cross_dif(x2q, x1q, x1p, cross[lv], crw[lv], heads, y2p, y2q, L, C);
    { float* t = x1p; x1p = y1p; y1p = t; t = x2p; x2p = y2p; y2p = t;
      unsigned short* u = x1q; x1q = y1q; y1q = u;
      u = x2q; x2q = y2q; y2q = u; }

    for (int bb = 0; bb < DEPTH[lv]; ++bb) {
      nat_layer(x1p, x1q, blks[lv][bb], blw[lv][bb], heads, L, C, H, W);
      nat_layer(x2p, x2q, blks[lv][bb], blw[lv][bb], heads, L, C, H, W);
    }

    lnorm(x1p, ong[lv], onb[lv], outp + out_off, nullptr, L, C);
    out_off += (size_t)L * C;
    lnorm(x2p, ong[lv], onb[lv], outp + out_off, nullptr, L, C);
    out_off += (size_t)L * C;

    if (lv < 3) {
      int Ho = H / 2, Wo = W / 2, Co = 2 * C;
      int total = Ho * Wo * Co;
      conv3x3_s2<<<dim3((total + 255) / 256), 256, 0, stream>>>(
          x1p, C, H, W, 0, down[lv].w, nullptr, qkvb, Co);
      lnorm(qkvb, down[lv].g, down[lv].b, y1p, y1q, Ho * Wo, Co);
      conv3x3_s2<<<dim3((total + 255) / 256), 256, 0, stream>>>(
          x2p, C, H, W, 0, down[lv].w, nullptr, qkvb, Co);
      lnorm(qkvb, down[lv].g, down[lv].b, y2p, y2q, Ho * Wo, Co);
      { float* t = x1p; x1p = y1p; y1p = t; t = x2p; x2p = y2p; y2p = t;
        unsigned short* u = x1q; x1q = y1q; y1q = u;
        u = x2q; x2q = y2q; y2q = u; }
      H = Ho; W = Wo; C = Co;
    }
  }
}
